// PointNetSetAbstraction_77077483094164
// MI455X (gfx1250) — matrix operand order repaired and checked by reading, not compiled, not run on hardware
//
#include <hip/hip_runtime.h>

// ---------------------------------------------------------------------------
// PointNet++ Set Abstraction for MI455X (gfx1250, wave32, WMMA)
// B=8, N=8192, NPOINT=1024, NSAMPLE=32, radius=0.4
//   1) FPS (8 blocks, register-resident distances, wave32 argmax reduce)
//   2) ball query + grouping -> x0 (M x 32, zero-padded from 6 channels)
//   3) pack W0/W1/W2 into f16 WMMA B-fragment layout (once per launch)
//   4) GEMM pass A: y0 = x0 @ W0^T            (+ channel sum/sumsq stats)
//   5) GEMM pass B: y1 = relu(bn0(y0)) @ W1^T (+ stats)
//   6) GEMM pass C: stats of relu(bn1(y1)) @ W2^T (no store)
//   7) fused: recompute y2, bn2+relu, max over k=32 -> new_points
// Bias terms cancel exactly through BN (mean shift) and are dropped.
// All matmuls: v_wmma_f32_16x16x32_f16, fragments loaded as b128 vectors.
// ---------------------------------------------------------------------------

#define BATCH   8
#define NPTS    8192
#define NPOINT  1024
#define NSAMPLE 32
#define MROWS   (BATCH * NPOINT * NSAMPLE)   // 262144
#define RAD2    (0.4f * 0.4f)
#define BN_EPS  1e-5f

// workspace layout (in floats)
#define X0_OFF  ((size_t)0)                              // 262144 * 32
#define Y0_OFF  ((size_t)(X0_OFF + (size_t)MROWS * 32))  // 262144 * 64
#define Y1_OFF  ((size_t)(Y0_OFF + (size_t)MROWS * 64))
#define ST_OFF  ((size_t)(Y1_OFF + (size_t)MROWS * 64))
// stats region: 12 slots of 128 floats
#define SUM0 0
#define SSQ0 128
#define SUM1 256
#define SSQ1 384
#define SUM2 512
#define SSQ2 640
#define SC0  768
#define SH0  896
#define SC1  1024
#define SH1  1152
#define SC2  1280
#define SH2  1408
#define ST_FLOATS 1536
// packed-weight region (in halfs, 32B-aligned): L0 2048, L1 4096, L2 8192
#define WP_OFF  ((size_t)(ST_OFF + ST_FLOATS))           // float offset
#define WP0_H   0
#define WP1_H   2048
#define WP2_H   6144
#define WP_FLOATS 7168

typedef __attribute__((ext_vector_type(16))) _Float16 v16h;
typedef __attribute__((ext_vector_type(8)))  float    v8f;

// ---------------------------------------------------------------------------
// WMMA helper: D = A(16x32 f16) * B(32x16 f16) + C(16x16 f32)
// ---------------------------------------------------------------------------
__device__ inline v8f wmma16(v16h a, v16h b, v8f c) {
  return __builtin_amdgcn_wmma_f32_16x16x32_f16(
      /*neg_a=*/false, a, /*neg_b=*/false, b,
      /*c_mod=*/(short)0, c, /*reuse_a=*/false, /*reuse_b=*/false);
}

__device__ inline float4 bn4(float4 v, float4 s, float4 h) {
  float4 r;
  r.x = fmaxf(fmaf(v.x, s.x, h.x), 0.f);
  r.y = fmaxf(fmaf(v.y, s.y, h.y), 0.f);
  r.z = fmaxf(fmaf(v.z, s.z, h.z), 0.f);
  r.w = fmaxf(fmaf(v.w, s.w, h.w), 0.f);
  return r;
}

__device__ inline v16h cvt_frag(float4 u0, float4 u1, float4 w0, float4 w1) {
  v16h a;
  a[0]  = (_Float16)u0.x; a[1]  = (_Float16)u0.y;
  a[2]  = (_Float16)u0.z; a[3]  = (_Float16)u0.w;
  a[4]  = (_Float16)u1.x; a[5]  = (_Float16)u1.y;
  a[6]  = (_Float16)u1.z; a[7]  = (_Float16)u1.w;
  a[8]  = (_Float16)w0.x; a[9]  = (_Float16)w0.y;
  a[10] = (_Float16)w0.z; a[11] = (_Float16)w0.w;
  a[12] = (_Float16)w1.x; a[13] = (_Float16)w1.y;
  a[14] = (_Float16)w1.z; a[15] = (_Float16)w1.w;
  return a;
}

// A fragment: 16x32 f16 per ISA table. lanes<16: K={0..7,16..23}+f*32,
// lanes>=16: K={8..15,24..31}+f*32. Two contiguous 8-float runs -> b128 loads.
// Optional fused BN(scale/shift per input channel)+ReLU (compile-time flag).
template <bool HASBN>
__device__ inline v16h load_a_frag(const float* __restrict__ rowp, int f,
                                   int lane, const float* __restrict__ bnsc,
                                   const float* __restrict__ bnsh) {
  const int off = (lane >> 4) * 8;
  const int k0 = f * 32 + off;
  const int k1 = k0 + 16;
  float4 u0 = *(const float4*)(rowp + k0);
  float4 u1 = *(const float4*)(rowp + k0 + 4);
  float4 w0 = *(const float4*)(rowp + k1);
  float4 w1 = *(const float4*)(rowp + k1 + 4);
  if (HASBN) {
    u0 = bn4(u0, *(const float4*)(bnsc + k0), *(const float4*)(bnsh + k0));
    u1 = bn4(u1, *(const float4*)(bnsc + k0 + 4), *(const float4*)(bnsh + k0 + 4));
    w0 = bn4(w0, *(const float4*)(bnsc + k1), *(const float4*)(bnsh + k1));
    w1 = bn4(w1, *(const float4*)(bnsc + k1 + 4), *(const float4*)(bnsh + k1 + 4));
  }
  return cvt_frag(u0, u1, w0, w1);
}

// Packed B fragment: one aligned v16h load per (t, f).
__device__ inline v16h load_b_packed(const _Float16* __restrict__ wp,
                                     int frag, int lane) {
  return ((const v16h*)wp)[frag * 32 + lane];
}

// ---------------------------------------------------------------------------
// Kernel: pack weights W[nout][kcw] fp32 -> f16 B-fragment layout.
// B frag layout: lane (l&15)=column n=t*16+(l&15); lanes<16 K={0..7,16..23},
// lanes>=16 K={8..15,24..31} (+f*32). Element i -> e=i&15, lane=(i>>4)&31, frag=i>>9.
// ---------------------------------------------------------------------------
__global__ void k_pack_w(const float* __restrict__ W, int kcw, int kfrags,
                         _Float16* __restrict__ wp, int total) {
  int i = blockIdx.x * blockDim.x + threadIdx.x;
  if (i >= total) return;
  int e    = i & 15;
  int lane = (i >> 4) & 31;
  int frag = i >> 9;
  int f = frag % kfrags;
  int t = frag / kfrags;
  int n = t * 16 + (lane & 15);
  int k = f * 32 + 8 * (lane >> 4) + (e & 7) + 16 * (e >> 3);
  float v = (k < kcw) ? W[(size_t)n * kcw + k] : 0.f;
  wp[i] = (_Float16)v;
}

// ---------------------------------------------------------------------------
// Kernel: zero stats region
// ---------------------------------------------------------------------------
__global__ void k_zero_stats(float* __restrict__ st) {
  int i = blockIdx.x * blockDim.x + threadIdx.x;
  if (i < ST_FLOATS) st[i] = 0.f;
}

// ---------------------------------------------------------------------------
// Kernel: farthest point sampling. 1 block per batch, 1024 threads,
// 8 points per thread in registers. Writes new_xyz to d_out.
// ---------------------------------------------------------------------------
__global__ __launch_bounds__(1024) void k_fps(const float* __restrict__ xyz,
                                              float* __restrict__ new_xyz) {
  __shared__ float s_cent[3];
  __shared__ float s_val[32];
  __shared__ int   s_idx[32];
  __shared__ int   s_far;

  const int b    = blockIdx.x;
  const int tid  = threadIdx.x;
  const int lane = tid & 31;
  const int wid  = tid >> 5;
  const float* xb = xyz + (size_t)b * NPTS * 3;

  float px[8], py[8], pz[8], d[8];
#pragma unroll
  for (int j = 0; j < 8; ++j) {
    int n = tid + j * 1024;
    px[j] = xb[n * 3 + 0];
    py[j] = xb[n * 3 + 1];
    pz[j] = xb[n * 3 + 2];
    d[j]  = 1e10f;
  }

  int far = 0;
  for (int it = 0; it < NPOINT; ++it) {
#pragma unroll
    for (int j = 0; j < 8; ++j) {
      if (far == tid + j * 1024) {
        s_cent[0] = px[j]; s_cent[1] = py[j]; s_cent[2] = pz[j];
      }
    }
    __syncthreads();
    if (tid == 0) {
      float* o = new_xyz + ((size_t)b * NPOINT + it) * 3;
      o[0] = s_cent[0]; o[1] = s_cent[1]; o[2] = s_cent[2];
    }
    const float cx = s_cent[0], cy = s_cent[1], cz = s_cent[2];

    float best = -1.f;
    int   bi   = 0;
#pragma unroll
    for (int j = 0; j < 8; ++j) {
      float dx = px[j] - cx, dy = py[j] - cy, dz = pz[j] - cz;
      float dd = dx * dx + dy * dy + dz * dz;
      d[j] = fminf(d[j], dd);
      if (d[j] > best) { best = d[j]; bi = tid + j * 1024; }
    }
#pragma unroll
    for (int o = 16; o > 0; o >>= 1) {
      float ov = __shfl_xor(best, o);
      int   oi = __shfl_xor(bi, o);
      if (ov > best || (ov == best && oi < bi)) { best = ov; bi = oi; }
    }
    if (lane == 0) { s_val[wid] = best; s_idx[wid] = bi; }
    __syncthreads();
    if (tid < 32) {
      float v = s_val[tid];
      int   i = s_idx[tid];
#pragma unroll
      for (int o = 16; o > 0; o >>= 1) {
        float ov = __shfl_xor(v, o);
        int   oi = __shfl_xor(i, o);
        if (ov > v || (ov == v && oi < i)) { v = ov; i = oi; }
      }
      if (tid == 0) s_far = i;
    }
    __syncthreads();
    far = s_far;
  }
}

// ---------------------------------------------------------------------------
// Kernel: ball query + grouping. One wave per (b, s). Picks the 32 smallest
// in-radius indices (== reference top_k semantics), fills short groups with
// the first in-ball point, writes 6-ch feature zero-padded to 32 cols.
// ---------------------------------------------------------------------------
__device__ inline void write_row(float* __restrict__ row, float dx, float dy,
                                 float dz, const float* __restrict__ p) {
  float4* r4 = (float4*)row;
  r4[0] = make_float4(dx, dy, dz, p[0]);
  r4[1] = make_float4(p[1], p[2], 0.f, 0.f);
  float4 z = make_float4(0.f, 0.f, 0.f, 0.f);
  r4[2] = z; r4[3] = z; r4[4] = z; r4[5] = z; r4[6] = z; r4[7] = z;
}

__global__ __launch_bounds__(256) void k_ball_group(
    const float* __restrict__ xyz, const float* __restrict__ pts,
    const float* __restrict__ new_xyz, float* __restrict__ x0) {
  const int lane = threadIdx.x & 31;
  const int gs   = blockIdx.x * 8 + (threadIdx.x >> 5);  // 0..8191
  const int b    = gs >> 10;
  const float* xb = xyz + (size_t)b * NPTS * 3;
  const float* pb = pts + (size_t)b * NPTS * 3;
  const float cx = new_xyz[(size_t)gs * 3 + 0];
  const float cy = new_xyz[(size_t)gs * 3 + 1];
  const float cz = new_xyz[(size_t)gs * 3 + 2];
  float* rowbase = x0 + (size_t)gs * NSAMPLE * 32;

  int cnt = 0, first = -1;
  for (int base = 0; base < NPTS && cnt < NSAMPLE; base += 32) {
    int j = base + lane;
    float dx = xb[j * 3 + 0] - cx;
    float dy = xb[j * 3 + 1] - cy;
    float dz = xb[j * 3 + 2] - cz;
    bool pred = (dx * dx + dy * dy + dz * dz) <= RAD2;
    unsigned mask = (unsigned)__ballot(pred);
    if (first < 0 && mask) first = base + (__ffs(mask) - 1);
    int rank = __popc(mask & ((1u << lane) - 1u));
    int slot = cnt + rank;
    if (pred && slot < NSAMPLE)
      write_row(rowbase + (size_t)slot * 32, dx, dy, dz, pb + (size_t)j * 3);
    cnt += __popc(mask);
  }
  if (cnt < NSAMPLE) {
    if (first < 0) first = 0;  // unreachable (center is in-ball), safety
    int j = first;
    float dx = xb[j * 3 + 0] - cx;
    float dy = xb[j * 3 + 1] - cy;
    float dz = xb[j * 3 + 2] - cz;
    for (int slot = cnt + lane; slot < NSAMPLE; slot += 32)
      write_row(rowbase + (size_t)slot * 32, dx, dy, dz, pb + (size_t)j * 3);
  }
}

// ---------------------------------------------------------------------------
// Kernel: WMMA GEMM pass (fully unrolled via templates).
// One wave per 16-row tile of M. Optional fused input BN+ReLU, optional
// fp32 output store, optional per-channel sum/sumsq stats
// (LDS ds_add_f32 atomics -> one global atomic per channel per block).
// ---------------------------------------------------------------------------
template <int KCA, int KFRAGS, int NTILES, bool HASBN, bool STORE, bool STATS>
__global__ __launch_bounds__(256) void k_gemm(
    const float* __restrict__ in, const _Float16* __restrict__ wp,
    const float* __restrict__ bnsc, const float* __restrict__ bnsh,
    float* __restrict__ out, float* __restrict__ ssum,
    float* __restrict__ sssq) {
  __shared__ float lsum[NTILES * 16];
  __shared__ float lssq[NTILES * 16];
  const int tid  = threadIdx.x;
  const int lane = tid & 31;
  if (STATS && tid < NTILES * 16) { lsum[tid] = 0.f; lssq[tid] = 0.f; }
  __syncthreads();

  const int w  = blockIdx.x * 8 + (tid >> 5);  // row-tile id
  const int nl = lane & 15, hi = lane >> 4;
  const float* rowp = in + ((size_t)w * 16 + nl) * KCA;

  v16h afr[KFRAGS];
#pragma unroll
  for (int f = 0; f < KFRAGS; ++f)
    afr[f] = load_a_frag<HASBN>(rowp, f, lane, bnsc, bnsh);

#pragma unroll
  for (int t = 0; t < NTILES; ++t) {
    v8f acc = {0.f, 0.f, 0.f, 0.f, 0.f, 0.f, 0.f, 0.f};
#pragma unroll
    for (int f = 0; f < KFRAGS; ++f) {
      v16h bf = load_b_packed(wp, t * KFRAGS + f, lane);
      acc = wmma16(afr[f], bf, acc);
    }
    const int col = t * 16 + nl;
    if (STORE) {
      size_t rb = (size_t)w * 16 + hi * 8;
#pragma unroll
      for (int r = 0; r < 8; ++r)
        out[(rb + r) * (NTILES * 16) + col] = acc[r];
    }
    if (STATS) {
      float s = 0.f, q = 0.f;
#pragma unroll
      for (int r = 0; r < 8; ++r) { float v = acc[r]; s += v; q += v * v; }
      atomicAdd(&lsum[col], s);
      atomicAdd(&lssq[col], q);
    }
  }
  if (STATS) {
    __syncthreads();
    if (tid < NTILES * 16) {
      atomicAdd(&ssum[tid], lsum[tid]);
      atomicAdd(&sssq[tid], lssq[tid]);
    }
  }
}

// ---------------------------------------------------------------------------
// Kernel: fold sum/sumsq into BN scale/shift.  a = relu(scale*y + shift)
// ---------------------------------------------------------------------------
__global__ void k_bn_finalize(const float* __restrict__ sum,
                              const float* __restrict__ ssq,
                              const float* __restrict__ g,
                              const float* __restrict__ be,
                              float* __restrict__ sc, float* __restrict__ sh,
                              int nc) {
  int c = threadIdx.x;
  if (c < nc) {
    const float inv = 1.f / (float)MROWS;
    float m = sum[c] * inv;
    float v = ssq[c] * inv - m * m;
    float s = g[c] * rsqrtf(v + BN_EPS);
    sc[c] = s;
    sh[c] = be[c] - m * s;
  }
}

// ---------------------------------------------------------------------------
// Kernel: fused layer-2 recompute + BN2 + ReLU + max over k=32.
// One wave per group (two 16-row tiles x 8 column tiles, nout=128).
// ---------------------------------------------------------------------------
__global__ __launch_bounds__(256) void k_mlp2_maxpool(
    const float* __restrict__ y1, const _Float16* __restrict__ wp2,
    const float* __restrict__ sc1, const float* __restrict__ sh1,
    const float* __restrict__ sc2, const float* __restrict__ sh2,
    float* __restrict__ outp) {
  const int lane = threadIdx.x & 31;
  const int g    = blockIdx.x * 8 + (threadIdx.x >> 5);  // 0..8191
  const int nl   = lane & 15;
  const size_t m0 = (size_t)g * NSAMPLE;
  const float* r0 = y1 + (m0 + nl) * 64;
  const float* r1 = y1 + (m0 + 16 + nl) * 64;

  v16h a0[2], a1[2];
  a0[0] = load_a_frag<true>(r0, 0, lane, sc1, sh1);
  a0[1] = load_a_frag<true>(r0, 1, lane, sc1, sh1);
  a1[0] = load_a_frag<true>(r1, 0, lane, sc1, sh1);
  a1[1] = load_a_frag<true>(r1, 1, lane, sc1, sh1);

#pragma unroll
  for (int t = 0; t < 8; ++t) {
    v8f acc0 = {0.f, 0.f, 0.f, 0.f, 0.f, 0.f, 0.f, 0.f};
    v8f acc1 = {0.f, 0.f, 0.f, 0.f, 0.f, 0.f, 0.f, 0.f};
    v16h b0 = load_b_packed(wp2, t * 2 + 0, lane);
    v16h b1 = load_b_packed(wp2, t * 2 + 1, lane);
    acc0 = wmma16(a0[0], b0, acc0);
    acc0 = wmma16(a0[1], b1, acc0);
    acc1 = wmma16(a1[0], b0, acc1);
    acc1 = wmma16(a1[1], b1, acc1);

    const int ch = t * 16 + nl;
    const float sc = sc2[ch], sh = sh2[ch];
    float mx = 0.f;  // relu output >= 0
#pragma unroll
    for (int r = 0; r < 8; ++r) {
      mx = fmaxf(mx, fmaxf(fmaf(acc0[r], sc, sh), 0.f));
      mx = fmaxf(mx, fmaxf(fmaf(acc1[r], sc, sh), 0.f));
    }
    mx = fmaxf(mx, __shfl_xor(mx, 16));
    if (lane < 16) outp[(size_t)g * 128 + ch] = mx;
  }
}

// ---------------------------------------------------------------------------
// Host launch
// ---------------------------------------------------------------------------
extern "C" void kernel_launch(void* const* d_in, const int* in_sizes, int n_in,
                              void* d_out, int out_size, void* d_ws,
                              size_t ws_size, hipStream_t stream) {
  const float* xyz = (const float*)d_in[0];
  const float* pts = (const float*)d_in[1];
  const float* W0  = (const float*)d_in[2];
  const float* g0  = (const float*)d_in[4];
  const float* be0 = (const float*)d_in[5];
  const float* W1  = (const float*)d_in[6];
  const float* g1  = (const float*)d_in[8];
  const float* be1 = (const float*)d_in[9];
  const float* W2  = (const float*)d_in[10];
  const float* g2  = (const float*)d_in[12];
  const float* be2 = (const float*)d_in[13];

  float* out     = (float*)d_out;
  float* new_xyz = out;  // 8*1024*3
  float* new_pts = out + (size_t)BATCH * NPOINT * 3;

  float* ws = (float*)d_ws;
  float* x0 = ws + X0_OFF;
  float* y0 = ws + Y0_OFF;
  float* y1 = ws + Y1_OFF;
  float* st = ws + ST_OFF;
  _Float16* wpb = (_Float16*)(ws + WP_OFF);
  _Float16* wp0 = wpb + WP0_H;
  _Float16* wp1 = wpb + WP1_H;
  _Float16* wp2 = wpb + WP2_H;

  k_zero_stats<<<(ST_FLOATS + 255) / 256, 256, 0, stream>>>(st);
  k_pack_w<<<(2048 + 255) / 256, 256, 0, stream>>>(W0, 6, 1, wp0, 2048);
  k_pack_w<<<(4096 + 255) / 256, 256, 0, stream>>>(W1, 64, 2, wp1, 4096);
  k_pack_w<<<(8192 + 255) / 256, 256, 0, stream>>>(W2, 64, 2, wp2, 8192);

  k_fps<<<BATCH, 1024, 0, stream>>>(xyz, new_xyz);
  k_ball_group<<<(BATCH * NPOINT) / 8, 256, 0, stream>>>(xyz, pts, new_xyz, x0);

  const int tiles_grid = (MROWS / 16) / 8;  // 2048 blocks, 8 waves each

  // layer 0: x0 (K=32 padded) -> y0 (64 ch), stats
  k_gemm<32, 1, 4, false, true, true><<<tiles_grid, 256, 0, stream>>>(
      x0, wp0, nullptr, nullptr, y0, st + SUM0, st + SSQ0);
  k_bn_finalize<<<1, 128, 0, stream>>>(st + SUM0, st + SSQ0, g0, be0,
                                       st + SC0, st + SH0, 64);
  // layer 1: bn0+relu(y0) -> y1 (64 ch), stats
  k_gemm<64, 2, 4, true, true, true><<<tiles_grid, 256, 0, stream>>>(
      y0, wp1, st + SC0, st + SH0, y1, st + SUM1, st + SSQ1);
  k_bn_finalize<<<1, 128, 0, stream>>>(st + SUM1, st + SSQ1, g1, be1,
                                       st + SC1, st + SH1, 64);
  // layer 2 stats only (y2 not materialized)
  k_gemm<64, 2, 8, true, false, true><<<tiles_grid, 256, 0, stream>>>(
      y1, wp2, st + SC1, st + SH1, nullptr, st + SUM2, st + SSQ2);
  k_bn_finalize<<<1, 128, 0, stream>>>(st + SUM2, st + SSQ2, g2, be2,
                                       st + SC2, st + SH2, 128);
  // fused recompute + bn2 + relu + maxpool
  k_mlp2_maxpool<<<(BATCH * NPOINT) / 8, 256, 0, stream>>>(
      y1, wp2, st + SC1, st + SH1, st + SC2, st + SH2, new_pts);
}
